// LSTMAutoEncoder_34213709480556
// MI455X (gfx1250) — matrix operand order repaired and checked by reading, not compiled, not run on hardware
//
#include <hip/hip_runtime.h>
#include <hip/hip_bf16.h>

// ---------------------------------------------------------------------------
// LSTM autoencoder, persistent-kernel implementation for gfx1250 (MI455X).
//   B=128, S=512, D=128, H=1024, 4H=4096.
// Round-3 changes vs round-2:
//   * Decoder y-projection (y = h @ out_W^T + out_b, 128x128, K=1024) now uses
//     ALL 64 WGs: 8x8 grid of 16x16 tiles, split-K across the 8 waves of each
//     WG (K=128 chunk per wave, 4 WMMAs) + LDS cross-wave reduction.
//     Previously 8 WGs worked while 56 idled at the barrier.
// Carried over:
//   * 64 persistent WGs, 16 hidden units each -> 4 acc tiles/wave (no spills).
//   * Weight slices staged in LDS once per phase (loop-invariant over 512
//     steps); inner loop = global_load_b128 (A) + ds_load_b128 (B) -> WMMA.
//   * Grid-wide ticket barrier, counter reset by same-stream init kernel.
//   * Decoder feedback y is zero-padded to H -> only first 128 K-cols of
//     dec_Wih ever matter after t=0.
// ---------------------------------------------------------------------------

#define NWG 64
#define TPB 256
#define B_  128
#define S_  512
#define D_  128
#define H_  1024
#define G4  4096   // 4*H
#define JW  16     // hidden units per workgroup

typedef __attribute__((ext_vector_type(16))) __bf16         v16bf;
typedef __attribute__((ext_vector_type(8)))  float          v8f;
typedef __attribute__((ext_vector_type(8)))  unsigned short v8us;

union Frag32 { v16bf bf; v8us us[2]; };

__device__ __forceinline__ v16bf load_frag(const unsigned short* p0,
                                           const unsigned short* p1) {
  Frag32 f;
  f.us[0] = *reinterpret_cast<const v8us*>(p0);
  f.us[1] = *reinterpret_cast<const v8us*>(p1);
  return f.bf;
}

__device__ __forceinline__ unsigned short f2bf(float f) {
  union { float f; unsigned u; } v; v.f = f;
  unsigned r = v.u + 0x7FFFu + ((v.u >> 16) & 1u);   // round-to-nearest-even
  return (unsigned short)(r >> 16);
}

__device__ __forceinline__ float sigm(float x)      { return 1.f / (1.f + __expf(-x)); }
__device__ __forceinline__ float tanh_fast(float x) { return 1.f - 2.f / (1.f + __expf(2.f * x)); }

// Grid-wide barrier; cnt starts at 0 (set by k_setup on the same stream).
__device__ __forceinline__ void grid_barrier(unsigned* cnt) {
  __syncthreads();
  if (threadIdx.x == 0) {
    unsigned ticket = __hip_atomic_fetch_add(cnt, 1u, __ATOMIC_ACQ_REL,
                                             __HIP_MEMORY_SCOPE_AGENT);
    unsigned target = (ticket / NWG + 1u) * NWG;
    while (__hip_atomic_load(cnt, __ATOMIC_ACQUIRE,
                             __HIP_MEMORY_SCOPE_AGENT) < target) {
      __builtin_amdgcn_s_sleep(1);
    }
  }
  __syncthreads();
}

// Stage this WG's 64 gate-rows (4 gates x 16 units) of a weight matrix into
// LDS: lds row q = g*16+jj  <-  global row g*H + j0 + jj, cols [0, dstLen).
__device__ __forceinline__ void stage_rows(unsigned short* dst,
                                           const unsigned short* __restrict__ src,
                                           int j0, int dstLen, int srcStride) {
  const int vecPerRow = dstLen >> 3;          // v8us per row
  const int totalVec  = 64 * vecPerRow;
  for (int v = threadIdx.x; v < totalVec; v += TPB) {
    const int q  = v / vecPerRow;             // lds row 0..63
    const int c  = (v - q * vecPerRow) << 3;  // column
    const int g  = q >> 4, jj = q & 15;
    const v8us* s = reinterpret_cast<const v8us*>(
        src + (size_t)(g * H_ + j0 + jj) * srcStride + c);
    *reinterpret_cast<v8us*>(dst + (size_t)q * dstLen + c) = *s;
  }
}

// acc[g] += A(16x32 rows of aRow) * B^T, B rows (g*16+nl) from LDS.
__device__ __forceinline__ void gemm_lds(v8f (&acc)[4],
                                         const unsigned short* __restrict__ aRow,
                                         const unsigned short* ldsW,
                                         int rowLen, int K, int lane) {
  const int kh8  = (lane >> 4) << 3;
  const int kh16 = (lane >> 4) << 4;
  const int nl   = lane & 15;
  for (int k = 0; k < K; k += 32) {
    v16bf a = load_frag(aRow + k + kh16, aRow + k + kh16 + 8);
#pragma unroll
    for (int g = 0; g < 4; ++g) {
      const unsigned short* br = ldsW + (size_t)(g * 16 + nl) * rowLen + k + kh16;
      v16bf b = load_frag(br, br + 8);
      acc[g] = __builtin_amdgcn_wmma_f32_16x16x32_bf16(
          false, a, false, b, (short)0, acc[g], false, false);
    }
  }
}

// Same but B read from global (used once: decoder t=0 full-width input GEMM).
__device__ __forceinline__ void gemm_global(v8f (&acc)[4],
                                            const unsigned short* __restrict__ aRow,
                                            const unsigned short* __restrict__ Bm,
                                            int bStride, int j0, int K, int lane) {
  const int kh8  = (lane >> 4) << 3;
  const int kh16 = (lane >> 4) << 4;
  const int nl   = lane & 15;
  for (int k = 0; k < K; k += 32) {
    v16bf a = load_frag(aRow + k + kh16, aRow + k + kh16 + 8);
#pragma unroll
    for (int g = 0; g < 4; ++g) {
      const unsigned short* br =
          Bm + (size_t)(g * H_ + j0 + nl) * bStride + k + kh16;
      v16bf b = load_frag(br, br + 8);
      acc[g] = __builtin_amdgcn_wmma_f32_16x16x32_bf16(
          false, a, false, b, (short)0, acc[g], false, false);
    }
  }
}

// ------------------------------ init kernels -------------------------------

__global__ void k_cvt(const float* __restrict__ src,
                      unsigned short* __restrict__ dst, int n) {
  for (int i = blockIdx.x * blockDim.x + threadIdx.x; i < n;
       i += gridDim.x * blockDim.x)
    dst[i] = f2bf(src[i]);
}

__global__ void k_setup(const float* __restrict__ bih_e, const float* __restrict__ bhh_e,
                        const float* __restrict__ bih_d, const float* __restrict__ bhh_d,
                        const float* __restrict__ outb_in,
                        float* enc_b, float* dec_b, float* out_b,
                        unsigned short* hbuf, float* cbuf, unsigned* cnt) {
  const int i      = blockIdx.x * blockDim.x + threadIdx.x;
  const int stride = gridDim.x * blockDim.x;
  for (int k = i; k < G4; k += stride) {
    enc_b[k] = bih_e[k] + bhh_e[k];
    dec_b[k] = bih_d[k] + bhh_d[k];
  }
  for (int k = i; k < D_; k += stride) out_b[k] = outb_in[k];
  for (int k = i; k < 2 * B_ * H_; k += stride) hbuf[k] = 0;   // h ping-pong = 0
  for (int k = i; k < B_ * H_; k += stride) cbuf[k] = 0.f;     // c0 = 0
  if (i == 0) *cnt = 0u;
}

// ---------------------------- persistent kernel ----------------------------

// LDS: [ Wih slice 64x128 | Whh slice 64x1024 ] bf16 = 144 KB,
//      + 8x256 f32 y split-K reduction buffer (8 KB) + loss red (1 KB).
#define LDS_WIH_OFF 0
#define LDS_WHH_OFF (64 * 128)
__shared__ unsigned short s_w[64 * 128 + 64 * 1024];
__shared__ float s_yred[8 * 256];

__global__ __launch_bounds__(TPB, 1) void lstm_ae_persistent(
    const unsigned short* __restrict__ xbf,
    const unsigned short* __restrict__ encWih,   // [4096][128]
    const unsigned short* __restrict__ encWhh,   // [4096][1024]
    const unsigned short* __restrict__ decWih,   // [4096][1024]
    const unsigned short* __restrict__ decWhh,   // [4096][1024]
    const unsigned short* __restrict__ outW,     // [128][1024]
    const float* __restrict__ enc_b, const float* __restrict__ dec_b,
    const float* __restrict__ out_b,
    unsigned short* hbase,            // 2 x [B][H] ping-pong (contiguous)
    float* cbuf,                      // [B][H]
    unsigned short* ybase,            // 2 x [B][D] ping-pong (contiguous)
    unsigned* cnt, float* partials,
    const float* __restrict__ x_f32,
    float* __restrict__ recon,        // [B][S][D]
    float* __restrict__ encoded,      // [B][S][H]
    float* __restrict__ loss_out) {
  const int wg   = blockIdx.x;            // 0..63, owns hidden units j0..j0+16
  const int lane = threadIdx.x & 31;
  const int wave = threadIdx.x >> 5;      // 0..7, owns batch rows m0..m0+16
  const int kh   = lane >> 4;             // C-tile: M = r + kh*8
  const int nl   = lane & 15;             // C-tile: N = nl
  const int m0   = wave * 16;
  const int j0   = wg * JW;
  const int kh8  = kh << 3, kh16 = kh << 4;
  const v8f vzero = {0.f, 0.f, 0.f, 0.f, 0.f, 0.f, 0.f, 0.f};

  // ------------------------------ encoder ------------------------------
  stage_rows(s_w + LDS_WIH_OFF, encWih, j0, D_, D_);
  stage_rows(s_w + LDS_WHH_OFF, encWhh, j0, H_, H_);
  __syncthreads();

  for (int t = 0; t < S_; ++t) {
    const unsigned short* hread  = hbase + (size_t)(t & 1) * (B_ * H_);
    unsigned short*       hwrite = hbase + (size_t)((t + 1) & 1) * (B_ * H_);
    v8f acc[4];
#pragma unroll
    for (int g = 0; g < 4; ++g) acc[g] = vzero;

    gemm_lds(acc, hread + (size_t)(m0 + nl) * H_, s_w + LDS_WHH_OFF, H_, H_, lane);
    gemm_lds(acc, xbf + ((size_t)(m0 + nl) * S_ + t) * D_, s_w + LDS_WIH_OFF, D_, D_, lane);

    {
      const int j = j0 + nl;
      const float bi = enc_b[0 * H_ + j], bf = enc_b[1 * H_ + j];
      const float bg = enc_b[2 * H_ + j], bo = enc_b[3 * H_ + j];
#pragma unroll
      for (int r = 0; r < 8; ++r) {
        const int m = m0 + r + kh * 8;
        const size_t ci = (size_t)m * H_ + j;
        const float iv = sigm(acc[0][r] + bi);
        const float fv = sigm(acc[1][r] + bf);
        const float gv = tanh_fast(acc[2][r] + bg);
        const float ov = sigm(acc[3][r] + bo);
        const float cn = fv * cbuf[ci] + iv * gv;
        const float hn = ov * tanh_fast(cn);
        cbuf[ci]   = cn;
        hwrite[ci] = f2bf(hn);
        encoded[(size_t)m * (S_ * H_) + (size_t)t * H_ + j] = hn;
      }
    }
    grid_barrier(cnt);
  }

  // ------------------------------ decoder ------------------------------
  // Re-stage weights: dec_Wih first 128 K-cols (zero-padded-y feedback only
  // touches those) and dec_Whh. t=0's full-width input GEMM reads global.
  stage_rows(s_w + LDS_WIH_OFF, decWih, j0, D_, H_);
  stage_rows(s_w + LDS_WHH_OFF, decWhh, j0, H_, H_);
  __syncthreads();

  // y-phase tiling: WG -> 16x16 tile (mtile = wg>>3, ntile = wg&7),
  // split-K across the 8 waves, reduced through s_yred.
  const int mtile = wg >> 3, ntile = wg & 7;

  for (int t = 0; t < S_; ++t) {
    const unsigned short* hread  = hbase + (size_t)(t & 1) * (B_ * H_);
    unsigned short*       hwrite = hbase + (size_t)((t + 1) & 1) * (B_ * H_);
    v8f acc[4];
#pragma unroll
    for (int g = 0; g < 4; ++g) acc[g] = vzero;

    gemm_lds(acc, hread + (size_t)(m0 + nl) * H_, s_w + LDS_WHH_OFF, H_, H_, lane);
    if (t == 0) {  // inp = h_n (full width) @ dec_Wih^T, from global
      gemm_global(acc, hbase + (size_t)(m0 + nl) * H_, decWih, H_, j0, H_, lane);
    } else {       // inp = pad(y): only first 128 K-cols matter
      const unsigned short* yread = ybase + (size_t)(t & 1) * (B_ * D_);
      gemm_lds(acc, yread + (size_t)(m0 + nl) * D_, s_w + LDS_WIH_OFF, D_, D_, lane);
    }

    {
      const int j = j0 + nl;
      const float bi = dec_b[0 * H_ + j], bf = dec_b[1 * H_ + j];
      const float bg = dec_b[2 * H_ + j], bo = dec_b[3 * H_ + j];
#pragma unroll
      for (int r = 0; r < 8; ++r) {
        const int m = m0 + r + kh * 8;
        const size_t ci = (size_t)m * H_ + j;
        const float iv = sigm(acc[0][r] + bi);
        const float fv = sigm(acc[1][r] + bf);
        const float gv = tanh_fast(acc[2][r] + bg);
        const float ov = sigm(acc[3][r] + bo);
        const float cn = fv * cbuf[ci] + iv * gv;
        const float hn = ov * tanh_fast(cn);
        cbuf[ci]   = cn;
        hwrite[ci] = f2bf(hn);
      }
    }
    grid_barrier(cnt);

    // ---- y = h_new @ out_W^T + out_b : one 16x16 tile per WG, split-K ----
    {
      const unsigned short* hn   = hbase + (size_t)((t + 1) & 1) * (B_ * H_);
      const unsigned short* aRow = hn + (size_t)(mtile * 16 + nl) * H_;
      const unsigned short* bRow = outW + (size_t)(ntile * 16 + nl) * H_;
      const int k0 = wave * 128;                 // this wave's K chunk
      v8f ya = vzero;
#pragma unroll
      for (int kk = 0; kk < 128; kk += 32) {
        const int k = k0 + kk;
        v16bf a = load_frag(aRow + k + kh16, aRow + k + kh16 + 8);
        v16bf b = load_frag(bRow + k + kh16, bRow + k + kh16 + 8);
        ya = __builtin_amdgcn_wmma_f32_16x16x32_bf16(false, a, false, b,
                                                     (short)0, ya, false, false);
      }
#pragma unroll
      for (int r = 0; r < 8; ++r) s_yred[wave * 256 + r * 32 + lane] = ya[r];
      __syncthreads();

      // one thread per tile element: reduce 8 wave partials
      const int e  = threadIdx.x;                // 0..255
      const int er = e >> 5, el = e & 31;        // stored (r, lane)
      float sum = 0.f;
#pragma unroll
      for (int w = 0; w < 8; ++w) sum += s_yred[w * 256 + e];
      const int m = mtile * 16 + er + ((el >> 4) << 3);
      const int n = ntile * 16 + (el & 15);
      const float yv = sum + out_b[n];
      recon[(size_t)m * (S_ * D_) + (size_t)t * D_ + n] = yv;
      unsigned short* ywrite = ybase + (size_t)((t + 1) & 1) * (B_ * D_);
      ywrite[m * D_ + n] = f2bf(yv);
    }
    grid_barrier(cnt);
  }

  // ----------------------------- recon loss ----------------------------
  __shared__ float red[TPB];
  float sacc = 0.f;
  const size_t total = (size_t)B_ * S_ * D_;
  for (size_t i = (size_t)wg * TPB + threadIdx.x; i < total;
       i += (size_t)NWG * TPB) {
    const float d = recon[i] - x_f32[i];
    sacc += d * d;
  }
  red[threadIdx.x] = sacc;
  __syncthreads();
  for (int off = TPB / 2; off > 0; off >>= 1) {
    if (threadIdx.x < off) red[threadIdx.x] += red[threadIdx.x + off];
    __syncthreads();
  }
  if (threadIdx.x == 0) partials[wg] = red[0];
  grid_barrier(cnt);
  if (wg == 0 && threadIdx.x == 0) {
    float tot = 0.f;
    for (int i = 0; i < NWG; ++i) tot += partials[i];
    *loss_out = tot / (float)B_;
  }
}

// ----------------------------- workspace map -------------------------------
static constexpr size_t OFF_EWIH = 0;                                    // 4096*128 bf16
static constexpr size_t SZ_EWIH  = (size_t)G4 * D_ * 2;
static constexpr size_t OFF_EWHH = OFF_EWIH + SZ_EWIH;                   // 4096*1024 bf16
static constexpr size_t SZ_WHH   = (size_t)G4 * H_ * 2;
static constexpr size_t OFF_DWIH = OFF_EWHH + SZ_WHH;
static constexpr size_t OFF_DWHH = OFF_DWIH + SZ_WHH;
static constexpr size_t OFF_OUTW = OFF_DWHH + SZ_WHH;                    // 128*1024 bf16
static constexpr size_t SZ_OUTW  = (size_t)D_ * H_ * 2;
static constexpr size_t OFF_XBF  = OFF_OUTW + SZ_OUTW;                   // B*S*D bf16
static constexpr size_t SZ_XBF   = (size_t)B_ * S_ * D_ * 2;
static constexpr size_t OFF_EB   = OFF_XBF + SZ_XBF;                     // 4096 f32
static constexpr size_t OFF_DB   = OFF_EB + (size_t)G4 * 4;
static constexpr size_t OFF_OB   = OFF_DB + (size_t)G4 * 4;              // 128 f32 (pad 512)
static constexpr size_t OFF_H0   = OFF_OB + 512;                         // 2 x B*H bf16
static constexpr size_t SZ_H     = (size_t)B_ * H_ * 2;
static constexpr size_t OFF_C    = OFF_H0 + 2 * SZ_H;                    // B*H f32
static constexpr size_t OFF_Y0   = OFF_C + (size_t)B_ * H_ * 4;          // 2 x B*D bf16
static constexpr size_t SZ_Y     = (size_t)B_ * D_ * 2;
static constexpr size_t OFF_CNT  = OFF_Y0 + 2 * SZ_Y;                    // barrier counter
static constexpr size_t OFF_PART = OFF_CNT + 256;                        // 64 f32 partials
// total ~44.4 MB of workspace

extern "C" void kernel_launch(void* const* d_in, const int* in_sizes, int n_in,
                              void* d_out, int out_size, void* d_ws, size_t ws_size,
                              hipStream_t stream) {
  (void)in_sizes; (void)n_in; (void)out_size; (void)ws_size;

  const float* x       = (const float*)d_in[0];
  const float* encWihF = (const float*)d_in[1];
  const float* encWhhF = (const float*)d_in[2];
  const float* enc_bih = (const float*)d_in[3];
  const float* enc_bhh = (const float*)d_in[4];
  const float* decWihF = (const float*)d_in[5];
  const float* decWhhF = (const float*)d_in[6];
  const float* dec_bih = (const float*)d_in[7];
  const float* dec_bhh = (const float*)d_in[8];
  const float* outWF   = (const float*)d_in[9];
  const float* outbF   = (const float*)d_in[10];

  char* ws = (char*)d_ws;
  unsigned short* wEWih = (unsigned short*)(ws + OFF_EWIH);
  unsigned short* wEWhh = (unsigned short*)(ws + OFF_EWHH);
  unsigned short* wDWih = (unsigned short*)(ws + OFF_DWIH);
  unsigned short* wDWhh = (unsigned short*)(ws + OFF_DWHH);
  unsigned short* wOutW = (unsigned short*)(ws + OFF_OUTW);
  unsigned short* wXbf  = (unsigned short*)(ws + OFF_XBF);
  float*          wEB   = (float*)(ws + OFF_EB);
  float*          wDB   = (float*)(ws + OFF_DB);
  float*          wOB   = (float*)(ws + OFF_OB);
  unsigned short* wH    = (unsigned short*)(ws + OFF_H0);
  float*          wC    = (float*)(ws + OFF_C);
  unsigned short* wY    = (unsigned short*)(ws + OFF_Y0);
  unsigned*       wCnt  = (unsigned*)(ws + OFF_CNT);
  float*          wPart = (float*)(ws + OFF_PART);

  float* out     = (float*)d_out;
  float* recon   = out;                                   // B*S*D
  float* encoded = out + (size_t)B_ * S_ * D_;            // B*S*H
  float* loss    = encoded + (size_t)B_ * S_ * H_;        // 1

  // fp32 -> bf16 weight/input conversion (same stream: ordered before main).
  k_cvt<<<512,  TPB, 0, stream>>>(encWihF, wEWih, G4 * D_);
  k_cvt<<<2048, TPB, 0, stream>>>(encWhhF, wEWhh, G4 * H_);
  k_cvt<<<2048, TPB, 0, stream>>>(decWihF, wDWih, G4 * H_);
  k_cvt<<<2048, TPB, 0, stream>>>(decWhhF, wDWhh, G4 * H_);
  k_cvt<<<256,  TPB, 0, stream>>>(outWF,   wOutW, D_ * H_);
  k_cvt<<<4096, TPB, 0, stream>>>(x,       wXbf,  B_ * S_ * D_);
  k_setup<<<256, TPB, 0, stream>>>(enc_bih, enc_bhh, dec_bih, dec_bhh, outbF,
                                   wEB, wDB, wOB, wH, wC, wCnt);

  lstm_ae_persistent<<<NWG, TPB, 0, stream>>>(
      wXbf, wEWih, wEWhh, wDWih, wDWhh, wOutW, wEB, wDB, wOB,
      wH, wC, wY, wCnt, wPart, x, recon, encoded, loss);
}
